// PureCartesianICTDTransformerLayer_29394756174148
// MI455X (gfx1250) — matrix operand order repaired and checked by reading, not compiled, not run on hardware
//
#include <hip/hip_runtime.h>
#include <hip/hip_bf16.h>
#include <math.h>

typedef __attribute__((ext_vector_type(16))) _Float16 f16x16;
typedef __attribute__((ext_vector_type(8)))  float    f32x8;

#define WMMA(a,b,c) __builtin_amdgcn_wmma_f32_16x16x32_f16(false,(a),false,(b),(short)0,(c),false,false)

// Force a global (addrspace 1) load: decouples weight streaming from DScnt
// (FLAT ops wait on both LOADcnt and DScnt; GLOBAL only on LOADcnt).
__device__ __forceinline__ float gload(const float* p) {
    return *(const __attribute__((address_space(1))) float*)p;
}

__device__ __forceinline__ void lds_fence() {
    asm volatile("s_wait_dscnt 0x0" ::: "memory");
}

// ---------------- path tables (LMAX = 2 -> 15 paths) ----------------
__device__ const int d_PL1[15] = {0,0,0,1,1,1,1,1,1,2,2,2,2,2,2};
__device__ const int d_PL2[15] = {0,1,2,0,1,1,1,2,2,0,1,1,2,2,2};
__device__ const int d_PLO[15] = {0,1,2,1,0,1,2,1,2,2,1,2,0,1,2};
__device__ const int d_LOFF[3] = {0,32,128};   // irrep block offsets in 288-wide feature
__device__ const int d_YOFF[3] = {0,1,4};      // offsets into per-edge y[9]
__device__ const int d_TOFF[3] = {0,64,256};   // offsets into 576-wide combined feature

// ---------------- WMMA fragment loaders (layouts per CDNA5 ISA) --------
// A: 16x32 f16, row-major in LDS with stride lda.
__device__ __forceinline__ f16x16 a_frag_lds(const _Float16* A, int lda, int k0, int lane) {
    int row = lane & 15;
    int kb  = k0 + ((lane & 16) ? 8 : 0);
    f16x16 a;
#pragma unroll
    for (int i = 0; i < 8; ++i) a[i]     = A[row*lda + kb + i];
#pragma unroll
    for (int i = 0; i < 8; ++i) a[8 + i] = A[row*lda + kb + 16 + i];
    return a;
}
// B: 32x16 f16, loaded from global f32 with arbitrary (kstride,nstride); OOB -> 0.
__device__ __forceinline__ f16x16 b_frag_g(const float* __restrict__ B, int kstride, int nstride,
                                           int k0, int n0, int kmax, int nmax, int lane) {
    int col = n0 + (lane & 15);
    int kb  = k0 + ((lane & 16) ? 8 : 0);
    bool cok = (col < nmax);
    f16x16 b;
#pragma unroll
    for (int i = 0; i < 16; ++i) {
        int k = kb + i + ((i >= 8) ? 8 : 0);
        float v = (cok && k < kmax) ? gload(B + k*kstride + col*nstride) : 0.0f;
        b[i] = (_Float16)v;
    }
    return b;
}

__device__ __forceinline__ float silu(float v) {
    // v * sigmoid(v) with v_rcp_f32 instead of IEEE divide
    return v * __builtin_amdgcn_rcpf(1.0f + __expf(-v));
}

// ---------------- CG init (device replica of _su2_cg / _q / _real_cg) -------------
struct dc { double re, im; };
__device__ __forceinline__ dc dcmul(dc a, dc b){ return {a.re*b.re - a.im*b.im, a.re*b.im + a.im*b.re}; }
__device__ __forceinline__ dc dcconj(dc a){ return {a.re, -a.im}; }
__device__ double dfact(int n){ double r = 1.0; for (int i = 2; i <= n; ++i) r *= (double)i; return r; }

__device__ void build_q(int l, dc q[5][5]) {
    for (int a = 0; a < 5; ++a) for (int b = 0; b < 5; ++b) q[a][b] = {0.0, 0.0};
    double s = 1.0 / sqrt(2.0);
    for (int m = -l; m < 0; ++m) { q[l+m][l-m] = {s, 0.0}; q[l+m][l+m] = {0.0, -s}; }
    q[l][l] = {1.0, 0.0};
    for (int m = 1; m <= l; ++m) {
        double sg = (m & 1) ? -1.0 : 1.0;
        q[l+m][l+m] = {sg*s, 0.0}; q[l+m][l-m] = {0.0, sg*s};
    }
    dc f = {1.0, 0.0}; dc mi = {0.0, -1.0};
    for (int t = 0; t < l; ++t) f = dcmul(f, mi);
    for (int a = 0; a < 2*l+1; ++a) for (int b = 0; b < 2*l+1; ++b) q[a][b] = dcmul(f, q[a][b]);
}

__device__ void su2_cg(int j1, int j2, int j3, double* C) { // C[5][5][5] flat
    for (int i = 0; i < 125; ++i) C[i] = 0.0;
    for (int m1 = -j1; m1 <= j1; ++m1)
      for (int m2 = -j2; m2 <= j2; ++m2) {
        int m3 = m1 + m2;
        if (m3 < -j3 || m3 > j3) continue;
        double pref = sqrt((2.0*j3+1.0)*dfact(j3+j1-j2)*dfact(j3-j1+j2)*dfact(j1+j2-j3)/dfact(j1+j2+j3+1));
        pref *= sqrt(dfact(j3+m3)*dfact(j3-m3)*dfact(j1-m1)*dfact(j1+m1)*dfact(j2-m2)*dfact(j2+m2));
        int kmin = 0; if (j2-j3-m1 > kmin) kmin = j2-j3-m1; if (j1+m2-j3 > kmin) kmin = j1+m2-j3;
        int kmax = j1+j2-j3; if (j1-m1 < kmax) kmax = j1-m1; if (j2+m2 < kmax) kmax = j2+m2;
        double s = 0.0;
        for (int k = kmin; k <= kmax; ++k) {
            double sg = (k & 1) ? -1.0 : 1.0;
            s += sg / (dfact(k)*dfact(j1+j2-j3-k)*dfact(j1-m1-k)*dfact(j2+m2-k)*dfact(j3-j2+m1+k)*dfact(j3-j1-m2+k));
        }
        C[((j1+m1)*5 + (j2+m2))*5 + (j3+m3)] = pref * s;
    }
}

// cg buffer: 15 paths x 125 dense floats, C2 at [1875]
__global__ void k_cg(float* __restrict__ cg) {
    if (threadIdx.x != 0 || blockIdx.x != 0) return;
    double su2[125]; double rr[125], ii[125];
    dc q1[5][5], q2[5][5], q3[5][5];
    for (int p = 0; p < 15; ++p) {
        int l1 = d_PL1[p], l2 = d_PL2[p], l3 = d_PLO[p];
        int dA = 2*l1+1, dB = 2*l2+1, dC = 2*l3+1;
        su2_cg(l1, l2, l3, su2);
        build_q(l1, q1); build_q(l2, q2); build_q(l3, q3);
        double mr = 0.0, mi = 0.0;
        for (int i = 0; i < dA; ++i)
          for (int j = 0; j < dB; ++j)
            for (int k = 0; k < dC; ++k) {
                dc acc = {0.0, 0.0};
                for (int a = 0; a < dA; ++a)
                  for (int b = 0; b < dB; ++b)
                    for (int c = 0; c < dC; ++c) {
                        double s = su2[(a*5+b)*5+c];
                        if (s == 0.0) continue;
                        dc t = dcmul(q1[a][i], q2[b][j]);
                        t = dcmul(t, dcconj(q3[c][k]));
                        acc.re += s * t.re; acc.im += s * t.im;
                    }
                int idx = (i*dB + j)*dC + k;
                rr[idx] = acc.re; ii[idx] = acc.im;
                if (fabs(acc.re) > mr) mr = fabs(acc.re);
                if (fabs(acc.im) > mi) mi = fabs(acc.im);
            }
        const double* src = (mr >= mi) ? rr : ii;
        for (int idx = 0; idx < dA*dB*dC; ++idx) cg[p*125 + idx] = (float)src[idx];
    }
    // C2 = sqrt(5 / sum_k (3*CG112[1,1,k])^2)
    const float* c6 = cg + 6*125; // path (1,1,2), dense [3][3][5]
    double ss = 0.0;
    for (int k = 0; k < 5; ++k) { double r = 3.0 * (double)c6[(1*3+1)*5 + k]; ss += r*r; }
    cg[1875] = (float)sqrt(5.0 / fmax(ss, 1e-12));
}

// ---------------- common edge geometry ----------------
__device__ void edge_geom(int e, const float* __restrict__ pos, const int* __restrict__ esrc,
                          const int* __restrict__ edst, const float* __restrict__ shifts,
                          const float* __restrict__ cell, const int* __restrict__ batch,
                          const float* __restrict__ cg, float* y9, float* plen, int* ps, int* pd) {
    int s = esrc[e], d = edst[e];
    int b = batch[s];
    const float* C  = cell + (size_t)b*9;
    const float* sh = shifts + (size_t)e*3;
    float sh0 = gload(sh+0), sh1 = gload(sh+1), sh2 = gload(sh+2);
    float v[3];
#pragma unroll
    for (int j = 0; j < 3; ++j) {
        float sv = sh0*gload(C+0*3+j) + sh1*gload(C+1*3+j) + sh2*gload(C+2*3+j);
        v[j] = gload(pos + (size_t)d*3+j) - gload(pos + (size_t)s*3+j) + sv;
    }
    float len = sqrtf(v[0]*v[0] + v[1]*v[1] + v[2]*v[2]);
    float inv = __builtin_amdgcn_rcpf(fmaxf(len, 1e-8f));
    float nx = v[0]*inv, ny = v[1]*inv, nz = v[2]*inv;
    const float SQ3 = 1.7320508075688772f;
    float y1[3] = {SQ3*ny, SQ3*nz, SQ3*nx};   // n[[1,2,0]]
    y9[0] = 1.0f; y9[1] = y1[0]; y9[2] = y1[1]; y9[3] = y1[2];
    float C2 = gload(cg + 1875);
    const float* cg112 = cg + 6*125;
#pragma unroll
    for (int k = 0; k < 5; ++k) {
        float a = 0.0f;
#pragma unroll
        for (int i = 0; i < 3; ++i)
#pragma unroll
            for (int j = 0; j < 3; ++j)
                a += y1[i]*y1[j]*gload(cg112 + (i*3+j)*5 + k);
        y9[4+k] = C2 * a;
    }
    *plen = len; *ps = s; *pd = d;
}

__device__ __forceinline__ void make_emb(float len, _Float16* dst /*32*/) {
#pragma unroll
    for (int i = 0; i < 16; ++i) {
        float v  = 5.0f * (float)(i+1) / 17.0f;
        float df = (len - v) * (17.0f/5.0f);
        dst[i] = (_Float16)(3.5714285714f * __expf(-df*df));   // (1/1.12)*sqrt(16)
    }
#pragma unroll
    for (int i = 16; i < 32; ++i) dst[i] = (_Float16)0.0f;
}

// ---------------- per-wave gate MLP: 16 edges, 16->64->64->15(16), all WMMA --------
__device__ void wave_gates_mlp(const _Float16* s_in /*[16][32]*/,
                               _Float16* s_h1, _Float16* s_h2 /*[16][64]*/,
                               float* s_g /*[16][16]*/,
                               const float* __restrict__ w1, const float* __restrict__ b1,
                               const float* __restrict__ w2, const float* __restrict__ b2,
                               const float* __restrict__ w3, const float* __restrict__ b3, int lane) {
    int col = lane & 15;
    int r0  = (lane & 16) ? 8 : 0;
    f16x16 a0 = a_frag_lds(s_in, 32, 0, lane);
#pragma unroll
    for (int n = 0; n < 4; ++n) {
        f32x8 acc = {0,0,0,0,0,0,0,0};
        f16x16 b = b_frag_g(w1, 64, 1, 0, n*16, 16, 64, lane);
        acc = WMMA(a0, b, acc);
        int c = n*16 + col; float bias = gload(b1 + c);
#pragma unroll
        for (int g = 0; g < 8; ++g) s_h1[(r0+g)*64 + c] = (_Float16)silu(acc[g] + bias);
    }
    lds_fence();
#pragma unroll
    for (int n = 0; n < 4; ++n) {
        f32x8 acc = {0,0,0,0,0,0,0,0};
#pragma unroll
        for (int ks = 0; ks < 64; ks += 32) {
            f16x16 a = a_frag_lds(s_h1, 64, ks, lane);
            f16x16 b = b_frag_g(w2, 64, 1, ks, n*16, 64, 64, lane);
            acc = WMMA(a, b, acc);
        }
        int c = n*16 + col; float bias = gload(b2 + c);
#pragma unroll
        for (int g = 0; g < 8; ++g) s_h2[(r0+g)*64 + c] = (_Float16)silu(acc[g] + bias);
    }
    lds_fence();
    {
        f32x8 acc = {0,0,0,0,0,0,0,0};
#pragma unroll
        for (int ks = 0; ks < 64; ks += 32) {
            f16x16 a = a_frag_lds(s_h2, 64, ks, lane);
            f16x16 b = b_frag_g(w3, 15, 1, ks, 0, 64, 15, lane);
            acc = WMMA(a, b, acc);
        }
        float bias = (col < 15) ? gload(b3 + col) : 0.0f;
#pragma unroll
        for (int g = 0; g < 8; ++g) s_g[(r0+g)*16 + col] = acc[g] + bias;
    }
    lds_fence();
}

// ---------------- kernel: zero ----------------
__global__ void k_zero(float* __restrict__ p, size_t n) {
    size_t i = (size_t)blockIdx.x * blockDim.x + threadIdx.x;
    size_t stride = (size_t)gridDim.x * blockDim.x;
    for (; i < n; i += stride) p[i] = 0.0f;
}

// ---------------- kernel: atom embedding MLP ----------------
__global__ void k_atom(const int* __restrict__ A, const float* __restrict__ emb,
                       const float* __restrict__ w1, const float* __restrict__ b1,
                       const float* __restrict__ w2, const float* __restrict__ b2,
                       float* __restrict__ ai, int N) {
    int n = blockIdx.x * blockDim.x + threadIdx.x;
    if (n >= N) return;
    const float* x = emb + (size_t)A[n]*16;
    float h[64];
    for (int j = 0; j < 64; ++j) {
        float a = gload(b1 + j);
        for (int i = 0; i < 16; ++i) a += gload(x + i)*gload(w1 + i*64+j);
        h[j] = silu(a);
    }
    for (int k = 0; k < 8; ++k) {
        float a = gload(b2 + k);
        for (int j = 0; j < 64; ++j) a += h[j]*gload(w2 + j*8+k);
        ai[(size_t)n*8 + k] = a;
    }
}

// ---------------- kernel: edge pass 1 (gates1 MLP + tp1, WMMA) ----------------
__global__ void __launch_bounds__(128) k_edge1(
        const float* __restrict__ pos, const int* __restrict__ batch,
        const int* __restrict__ esrc, const int* __restrict__ edst,
        const float* __restrict__ shifts, const float* __restrict__ cell,
        const float* __restrict__ ai, const float* __restrict__ tp1w,
        const float* __restrict__ w1, const float* __restrict__ b1,
        const float* __restrict__ w2, const float* __restrict__ b2,
        const float* __restrict__ w3, const float* __restrict__ b3,
        const float* __restrict__ cg, float* __restrict__ f1s,
        float* __restrict__ cnt, int E) {
    __shared__ _Float16 s_emb[128*32];
    __shared__ _Float16 s_ap [128*64];
    __shared__ float    s_y  [128*9];
    __shared__ int      s_dst[128];
    __shared__ _Float16 s_h1 [4*16*64];
    __shared__ _Float16 s_h2 [4*16*64];
    __shared__ float    s_g  [4*16*16];
    int tid = threadIdx.x, lane = tid & 31, wav = tid >> 5;
    int e0 = blockIdx.x * 128;
    int e  = e0 + tid;
    if (e < E) {
        float y9[9]; float len; int s, d;
        edge_geom(e, pos, esrc, edst, shifts, cell, batch, cg, y9, &len, &s, &d);
#pragma unroll
        for (int j = 0; j < 9; ++j) s_y[tid*9+j] = y9[j];
        s_dst[tid] = d;
        make_emb(len, s_emb + tid*32);
        const float* As = ai + (size_t)s*8;
        const float* Ad = ai + (size_t)d*8;
#pragma unroll
        for (int i = 0; i < 8; ++i) {
            float av = gload(As + i);
#pragma unroll
            for (int j = 0; j < 8; ++j) s_ap[tid*64 + i*8 + j] = (_Float16)(av * gload(Ad + j));
        }
        atomicAdd(&cnt[d], 1.0f);
    } else {
#pragma unroll
        for (int j = 0; j < 9; ++j) s_y[tid*9+j] = 0.0f;
        s_dst[tid] = 0;
        for (int i = 0; i < 32; ++i) s_emb[tid*32+i] = (_Float16)0.0f;
        for (int i = 0; i < 64; ++i) s_ap [tid*64+i] = (_Float16)0.0f;
    }
    __syncthreads();
    const int pidx[3] = {0, 3, 9};   // paths (0,0,0),(1,0,1),(2,0,2)
    int col = lane & 15, r0 = (lane & 16) ? 8 : 0;
    for (int grp = wav; grp < 8; grp += 4) {
        wave_gates_mlp(s_emb + grp*16*32, s_h1 + wav*1024, s_h2 + wav*1024,
                       s_g + wav*256, w1, b1, w2, b2, w3, b3, lane);
        f16x16 aA0  = a_frag_lds(s_ap + grp*16*64, 64, 0,  lane);
        f16x16 aA32 = a_frag_lds(s_ap + grp*16*64, 64, 32, lane);
#pragma unroll
        for (int p3 = 0; p3 < 3; ++p3) {
            int lo = p3, dim = 2*lo + 1;
            const float* W = tp1w + pidx[p3]*2048;   // [64 ij][32 o]
#pragma unroll
            for (int nt = 0; nt < 2; ++nt) {
                f32x8 acc = {0,0,0,0,0,0,0,0};
                f16x16 b0 = b_frag_g(W, 32, 1, 0,  nt*16, 64, 32, lane);
                acc = WMMA(aA0, b0, acc);
                f16x16 b1f = b_frag_g(W, 32, 1, 32, nt*16, 64, 32, lane);
                acc = WMMA(aA32, b1f, acc);
                int o = nt*16 + col;
#pragma unroll
                for (int g = 0; g < 8; ++g) {
                    int el = grp*16 + r0 + g;
                    float gate = s_g[wav*256 + (r0+g)*16 + pidx[p3]];
                    float sv = acc[g] * 0.125f * gate;   // norm = 1/sqrt(8*8)
                    float* base = f1s + (size_t)s_dst[el]*288 + d_LOFF[lo] + o*dim;
                    if (lo == 0) atomicAdd(base, sv);
                    else {
                        int yo = (lo == 1) ? 1 : 4;
                        for (int k = 0; k < dim; ++k)
                            atomicAdd(base + k, sv * s_y[el*9 + yo + k]);
                    }
                }
            }
        }
    }
}

// ---------------- kernel: divide by count ----------------
__global__ void k_div(float* __restrict__ f, const float* __restrict__ cnt, size_t total) {
    size_t i = (size_t)blockIdx.x * blockDim.x + threadIdx.x;
    if (i >= total) return;
    size_t n = i / 288;
    f[i] *= __builtin_amdgcn_rcpf(fmaxf(gload(cnt + n), 1.0f));
}

// ---------------- kernel: edge pass 2 (gates2 MLP + tp2, WMMA) ----------------
__global__ void __launch_bounds__(128) k_edge2(
        const float* __restrict__ pos, const int* __restrict__ batch,
        const int* __restrict__ esrc, const int* __restrict__ edst,
        const float* __restrict__ shifts, const float* __restrict__ cell,
        const float* __restrict__ f1, const float* __restrict__ tp2w,
        const float* __restrict__ w1, const float* __restrict__ b1,
        const float* __restrict__ w2, const float* __restrict__ b2,
        const float* __restrict__ w3, const float* __restrict__ b3,
        const float* __restrict__ cg, float* __restrict__ f2s, int E) {
    __shared__ _Float16 s_x1 [16*288];
    __shared__ float    s_y  [16*9];
    __shared__ float    s_g2 [16*16];
    __shared__ int      s_src[16];
    __shared__ int      s_dst[16];
    __shared__ _Float16 s_emb[16*32];
    __shared__ _Float16 s_h1 [16*64];
    __shared__ _Float16 s_h2 [16*64];
    __shared__ _Float16 s_u  [80*32];
    __shared__ float    s_out[16*288];
    int tid = threadIdx.x, lane = tid & 31, wav = tid >> 5;
    int e0 = blockIdx.x * 16;
    if (tid < 16) {
        int e = e0 + tid;
        if (e < E) {
            float y9[9]; float len; int s, d;
            edge_geom(e, pos, esrc, edst, shifts, cell, batch, cg, y9, &len, &s, &d);
            s_src[tid] = s; s_dst[tid] = d;
#pragma unroll
            for (int j = 0; j < 9; ++j) s_y[tid*9+j] = y9[j];
            make_emb(len, s_emb + tid*32);
        } else {
            s_src[tid] = 0; s_dst[tid] = 0;
#pragma unroll
            for (int j = 0; j < 9; ++j) s_y[tid*9+j] = 0.0f;
            for (int i = 0; i < 32; ++i) s_emb[tid*32+i] = (_Float16)0.0f;
        }
    }
    for (int i = tid; i < 16*288; i += 128) s_out[i] = 0.0f;
    __syncthreads();
    for (int i = tid; i < 16*288; i += 128) {
        int e = i / 288, c = i - e*288;
        s_x1[i] = (_Float16)gload(f1 + (size_t)s_src[e]*288 + c);
    }
    __syncthreads();
    if (wav == 0) wave_gates_mlp(s_emb, s_h1, s_h2, s_g2, w1, b1, w2, b2, w3, b3, lane);
    __syncthreads();
    int col = lane & 15, r0 = (lane & 16) ? 8 : 0;
    for (int p = 0; p < 15; ++p) {
        int l1 = d_PL1[p], l2 = d_PL2[p], lo = d_PLO[p];
        int dA = 2*l1+1, dB = 2*l2+1, dim = 2*lo+1;
        const float* cgp = cg + p*125;
        for (int pr = tid; pr < 512; pr += 128) {
            int e = pr >> 5, i = pr & 31;
            const _Float16* xr = s_x1 + e*288 + d_LOFF[l1] + i*dA;
            const float*    yr = s_y  + e*9   + d_YOFF[l2];
            for (int k = 0; k < dim; ++k) {
                float acc = 0.0f;
                for (int m = 0; m < dA; ++m) {
                    float xm = (float)xr[m];
                    for (int n2 = 0; n2 < dB; ++n2)
                        acc += xm * yr[n2] * gload(cgp + (m*dB + n2)*dim + k);
                }
                s_u[(e*dim + k)*32 + i] = (_Float16)acc;
            }
        }
        __syncthreads();
        int jobs = 2*dim;
        for (int j = wav; j < jobs; j += 4) {
            int rt = j >> 1, nt = j & 1;
            f32x8 acc = {0,0,0,0,0,0,0,0};
            f16x16 a = a_frag_lds(s_u + rt*16*32, 32, 0, lane);
            f16x16 b = b_frag_g(tp2w + p*1024, 32, 1, 0, nt*16, 32, 32, lane);
            acc = WMMA(a, b, acc);
            int o = nt*16 + col;
#pragma unroll
            for (int g = 0; g < 8; ++g) {
                int r = rt*16 + r0 + g;
                int e = r / dim, k = r - e*dim;
                float v = acc[g] * s_g2[e*16 + p] * 0.17677669529f;  // 1/sqrt(32)
                atomicAdd(&s_out[e*288 + d_LOFF[lo] + o*dim + k], v);
            }
        }
        __syncthreads();
    }
    for (int i = tid; i < 16*288; i += 128) {
        int e = i / 288, c = i - e*288;
        atomicAdd(&f2s[(size_t)s_dst[e]*288 + c], s_out[i]);
    }
}

// ---------------- generic WMMA block layer (16 rows, K->Nout) ----------------
__device__ void block_layer(const _Float16* s_in, int lda, int K,
                            const float* __restrict__ W, const float* __restrict__ B, int Nout,
                            _Float16* s_o, int ldo, bool act, int lane, int wav) {
    int col = lane & 15, r0 = (lane & 16) ? 8 : 0;
    int ctiles = (Nout + 15) / 16;
    for (int n = wav; n < ctiles; n += 4) {
        f32x8 acc = {0,0,0,0,0,0,0,0};
        for (int ks = 0; ks < K; ks += 32) {
            f16x16 a = a_frag_lds(s_in, lda, ks, lane);
            f16x16 b = b_frag_g(W, Nout, 1, ks, n*16, K, Nout, lane);
            acc = WMMA(a, b, acc);
        }
        int c = n*16 + col;
        if (c < Nout) {
            float bias = gload(B + c);
#pragma unroll
            for (int g = 0; g < 8; ++g) {
                float v = acc[g] + bias;
                if (act) v = silu(v);
                s_o[(r0+g)*ldo + c] = (_Float16)v;
            }
        }
    }
}
__device__ void block_layer_addf32(const _Float16* s_in, int lda, int K,
                                   const float* __restrict__ W, const float* __restrict__ B, int Nout,
                                   float* s_o, int ldo, int lane, int wav) {
    int col = lane & 15, r0 = (lane & 16) ? 8 : 0;
    int ctiles = (Nout + 15) / 16;
    for (int n = wav; n < ctiles; n += 4) {
        f32x8 acc = {0,0,0,0,0,0,0,0};
        for (int ks = 0; ks < K; ks += 32) {
            f16x16 a = a_frag_lds(s_in, lda, ks, lane);
            f16x16 b = b_frag_g(W, Nout, 1, ks, n*16, K, Nout, lane);
            acc = WMMA(a, b, acc);
        }
        int c = n*16 + col;
        if (c < Nout) {
            float bias = gload(B + c);
#pragma unroll
            for (int g = 0; g < 8; ++g) s_o[(r0+g)*ldo + c] += acc[g] + bias;
        }
    }
}

// ---------------- kernel: invariant readout ----------------
__global__ void __launch_bounds__(128) k_readout(
        const float* __restrict__ f1, const float* __restrict__ f2,
        const int* __restrict__ batch, const float* __restrict__ wread,
        const float* __restrict__ row1, const float* __restrict__ rob1,
        const float* __restrict__ row2, const float* __restrict__ rob2,
        const float* __restrict__ ptw1, const float* __restrict__ ptb1,
        const float* __restrict__ ptw2, const float* __restrict__ ptb2,
        const float* __restrict__ ptw3, const float* __restrict__ ptb3,
        const float* __restrict__ ptw4, const float* __restrict__ ptb4,
        const float* __restrict__ wsw, float* __restrict__ out, int N) {
    __shared__ _Float16 s_t   [16*576];
    __shared__ _Float16 s_gr  [16*64];
    __shared__ _Float16 s_feat[16*256];
    __shared__ _Float16 s_h   [16*128];
    __shared__ _Float16 s_hb  [16*128];
    __shared__ float    s_scal[16*32];
    __shared__ float    s_pred[16*32];
    int tid = threadIdx.x, lane = tid & 31, wav = tid >> 5;
    int n0 = blockIdx.x * 16;
    for (int i = tid; i < 16*576; i += 128) {
        int nd = i / 576, c = i - nd*576;
        int gn = n0 + nd;
        float v = 0.0f;
        if (gn < N) v = (c < 288) ? gload(f1 + (size_t)gn*288 + c)
                                  : gload(f2 + (size_t)gn*288 + (c-288));
        s_t[i] = (_Float16)v;
    }
    for (int i = tid; i < 512; i += 128) { s_scal[i] = 0.0f; s_pred[i] = 0.0f; }
    __syncthreads();
    // --- scalars: stream gram rows through WMMA against w_read ---
    f32x8 accS = {0,0,0,0,0,0,0,0};
    int col = lane & 15, r0 = (lane & 16) ? 8 : 0;
    for (int l = 0; l < 3; ++l) {
        int dim = 2*l + 1;
        float invs = rsqrtf((float)dim);
        for (int c = 0; c < 64; ++c) {
            for (int pr = tid; pr < 1024; pr += 128) {
                int nd = pr >> 6, dd = pr & 63;
                const _Float16* tc = s_t + nd*576 + d_TOFF[l] + c*dim;
                const _Float16* td = s_t + nd*576 + d_TOFF[l] + dd*dim;
                float a = 0.0f;
                for (int m = 0; m < dim; ++m) a += (float)tc[m] * (float)td[m];
                s_gr[nd*64 + dd] = (_Float16)(a * invs);
            }
            __syncthreads();
            if (wav < 2) {
                const float* B = wread + l*131072 + c*64;   // element(k=d,n=o) at n*4096 + k
#pragma unroll
                for (int ks = 0; ks < 64; ks += 32) {
                    f16x16 a = a_frag_lds(s_gr, 64, ks, lane);
                    f16x16 b = b_frag_g(B, 1, 4096, ks, wav*16, 64, 32, lane);
                    accS = WMMA(a, b, accS);
                }
            }
            __syncthreads();
        }
    }
    if (wav < 2) {
        int o = wav*16 + col;
#pragma unroll
        for (int g = 0; g < 8; ++g) s_scal[(r0+g)*32 + o] = accS[g];
    }
    __syncthreads();
    // --- feat = [scalars(32) | norms l0(64) | l1(64) | l2(64)] ---
    for (int pr = tid; pr < 512; pr += 128) {
        int nd = pr >> 5, o = pr & 31;
        s_feat[nd*256 + o] = (_Float16)s_scal[pr];
    }
    for (int pr = tid; pr < 16*192; pr += 128) {
        int nd = pr / 192, j = pr - nd*192;
        int l = j >> 6, cc = j & 63, dim = 2*l + 1;
        const _Float16* tc = s_t + nd*576 + d_TOFF[l] + cc*dim;
        float a = 1e-12f;
        for (int m = 0; m < dim; ++m) { float x = (float)tc[m]; a += x*x; }
        s_feat[nd*256 + 32 + j] = (_Float16)sqrtf(a);
    }
    __syncthreads();
    // --- node MLPs: ro(224->128->17) + pt(224->128->128->128->17) ---
    block_layer(s_feat, 256, 224, row1, rob1, 128, s_h, 128, true, lane, wav);
    __syncthreads();
    block_layer_addf32(s_h, 128, 128, row2, rob2, 17, s_pred, 32, lane, wav);
    __syncthreads();
    block_layer(s_feat, 256, 224, ptw1, ptb1, 128, s_h, 128, true, lane, wav);
    __syncthreads();
    block_layer(s_h, 128, 128, ptw2, ptb2, 128, s_hb, 128, true, lane, wav);
    __syncthreads();
    block_layer(s_hb, 128, 128, ptw3, ptb3, 128, s_h, 128, true, lane, wav);
    __syncthreads();
    block_layer_addf32(s_h, 128, 128, ptw4, ptb4, 17, s_pred, 32, lane, wav);
    __syncthreads();
    if (tid < 16 && (n0 + tid) < N) {
        float e = 0.0f;
        for (int j = 0; j < 17; ++j) e += s_pred[tid*32 + j] * wsw[j];
        atomicAdd(&out[batch[n0 + tid]], e);
    }
}

// ---------------- launcher ----------------
extern "C" void kernel_launch(void* const* d_in, const int* in_sizes, int n_in,
                              void* d_out, int out_size, void* d_ws, size_t ws_size,
                              hipStream_t stream) {
    const float* pos    = (const float*)d_in[0];
    const int*   A      = (const int*)  d_in[1];
    const int*   batch  = (const int*)  d_in[2];
    const int*   esrc   = (const int*)  d_in[3];
    const int*   edst   = (const int*)  d_in[4];
    const float* shifts = (const float*)d_in[5];
    const float* cell   = (const float*)d_in[6];
    const float* aemb   = (const float*)d_in[7];
    const float* amw1   = (const float*)d_in[8];
    const float* amb1   = (const float*)d_in[9];
    const float* amw2   = (const float*)d_in[10];
    const float* amb2   = (const float*)d_in[11];
    const float* tp1w   = (const float*)d_in[12];
    const float* f1w1   = (const float*)d_in[13];
    const float* f1b1   = (const float*)d_in[14];
    const float* f1w2   = (const float*)d_in[15];
    const float* f1b2   = (const float*)d_in[16];
    const float* f1w3   = (const float*)d_in[17];
    const float* f1b3   = (const float*)d_in[18];
    const float* tp2w   = (const float*)d_in[19];
    const float* f2w1   = (const float*)d_in[20];
    const float* f2b1   = (const float*)d_in[21];
    const float* f2w2   = (const float*)d_in[22];
    const float* f2b2   = (const float*)d_in[23];
    const float* f2w3   = (const float*)d_in[24];
    const float* f2b3   = (const float*)d_in[25];
    const float* wread  = (const float*)d_in[26];
    const float* row1   = (const float*)d_in[27];
    const float* rob1   = (const float*)d_in[28];
    const float* row2   = (const float*)d_in[29];
    const float* rob2   = (const float*)d_in[30];
    const float* wsw    = (const float*)d_in[31];
    const float* ptw1   = (const float*)d_in[32];
    const float* ptb1   = (const float*)d_in[33];
    const float* ptw2   = (const float*)d_in[34];
    const float* ptb2   = (const float*)d_in[35];
    const float* ptw3   = (const float*)d_in[36];
    const float* ptb3   = (const float*)d_in[37];
    const float* ptw4   = (const float*)d_in[38];
    const float* ptb4   = (const float*)d_in[39];

    int N = in_sizes[1];
    int E = in_sizes[3];

    float* ws   = (float*)d_ws;
    float* cg   = ws;                                  // 2048
    float* ai   = ws + 2048;                           // N*8
    float* f1   = ai + (size_t)N*8;                    // N*288
    float* f2   = f1 + (size_t)N*288;                  // N*288
    float* cnt  = f2 + (size_t)N*288;                  // N
    float* outF = (float*)d_out;

    size_t zn = (size_t)N*288*2 + (size_t)N;
    int zb = (int)((zn + 255) / 256);
    k_zero<<<zb, 256, 0, stream>>>(f1, zn);
    k_zero<<<1, 64, 0, stream>>>(outF, (size_t)out_size);
    k_cg<<<1, 1, 0, stream>>>(cg);
    k_atom<<<(N + 127)/128, 128, 0, stream>>>(A, aemb, amw1, amb1, amw2, amb2, ai, N);
    k_edge1<<<(E + 127)/128, 128, 0, stream>>>(pos, batch, esrc, edst, shifts, cell,
                                               ai, tp1w, f1w1, f1b1, f1w2, f1b2, f1w3, f1b3,
                                               cg, f1, cnt, E);
    {
        size_t t = (size_t)N*288;
        k_div<<<(int)((t + 255)/256), 256, 0, stream>>>(f1, cnt, t);
    }
    k_edge2<<<(E + 15)/16, 128, 0, stream>>>(pos, batch, esrc, edst, shifts, cell,
                                             f1, tp2w, f2w1, f2b1, f2w2, f2b2, f2w3, f2b3,
                                             cg, f2, E);
    {
        size_t t = (size_t)N*288;
        k_div<<<(int)((t + 255)/256), 256, 0, stream>>>(f2, cnt, t);
    }
    k_readout<<<(N + 15)/16, 128, 0, stream>>>(f1, f2, batch, wread,
                                               row1, rob1, row2, rob2,
                                               ptw1, ptb1, ptw2, ptb2, ptw3, ptb3, ptw4, ptb4,
                                               wsw, outF, N);
}
